// GraphNet_9259949490748
// MI455X (gfx1250) — matrix operand order repaired and checked by reading, not compiled, not run on hardware
//
#include <hip/hip_runtime.h>
#include <hip/hip_bf16.h>

typedef __attribute__((ext_vector_type(16))) _Float16 v16h;
typedef __attribute__((ext_vector_type(8)))  float    v8f;

#define NN 100000   // nodes
#define NE 1600000  // edges
#define NG 512      // graphs
#define NC 16       // classes

static inline int cdiv_h(long a, int b) { return (int)((a + b - 1) / b); }

// ---------------- utility kernels ----------------

__global__ void k_fill(float* __restrict__ p, float v, long n) {
  long i = (long)blockIdx.x * blockDim.x + threadIdx.x;
  if (i < n) p[i] = v;
}

__global__ void k_edge_deg(const long long* __restrict__ dst, float* __restrict__ deg, int E) {
  int e = blockIdx.x * blockDim.x + threadIdx.x;
  if (e < E) atomicAdd(&deg[(long)dst[e]], 1.0f);
}

__global__ void k_dinv(const float* __restrict__ deg, float* __restrict__ dinv, int n) {
  int i = blockIdx.x * blockDim.x + threadIdx.x;
  if (i < n) dinv[i] = rsqrtf(deg[i]);   // deg >= 1 always (self loop)
}

// ---------------- WMMA dense projection: H[M x NO] = X[M x KD] @ W[KD x NO] ----
// One wave computes one 16x16 output tile; K marched in steps of 32 (f16 WMMA).
// A layout (16-bit A 16x32, contiguous K, same as B): lanes 0-15 -> M=lane, elems: K = e;
// lanes 16-31 same M, K=16+e. B: lanes 0-15 -> N=lane, K=e; lanes 16-31 K=16+e.
// C/D f32: VGPR v -> row v (lanes 0-15) / v+8 (lanes 16-31), col = lane&15.
template<int KD, int NO>
__global__ void k_gemm_wmma(const float* __restrict__ X, const float* __restrict__ W,
                            float* __restrict__ H, int M) {
  const int lane = threadIdx.x & 31;
  const int wave = threadIdx.x >> 5;
  const int mt   = blockIdx.x * (blockDim.x >> 5) + wave;
  const int Mt   = M >> 4;                      // M is a multiple of 16 here
  if (mt >= Mt) return;                          // wave-uniform: EXEC stays all-ones
  const int  col0 = blockIdx.y << 4;
  const bool hiH  = (lane >= 16);
  const int  l15  = lane & 15;
  const long m    = (long)mt * 16 + l15;
  const int  n    = col0 + l15;

  v8f acc = {};
  for (int k0 = 0; k0 < KD; k0 += 32) {
    v16h a, b;
#pragma unroll
    for (int e = 0; e < 16; ++e) {
      int ka = k0 + e + (hiH ? 16 : 0);
      a[e] = (_Float16)((ka < KD) ? X[m * KD + ka] : 0.0f);
      int kb = k0 + e + (hiH ? 16 : 0);
      b[e] = (_Float16)((kb < KD) ? W[(long)kb * NO + n] : 0.0f);
    }
    acc = __builtin_amdgcn_wmma_f32_16x16x32_f16(
        /*neg_a=*/false, a, /*neg_b=*/false, b,
        /*c_mod=*/(short)0, acc, /*reuse_a=*/false, /*reuse_b=*/false);
  }
#pragma unroll
  for (int v = 0; v < 8; ++v) {
    long r = (long)mt * 16 + v + (hiH ? 8 : 0);
    H[r * NO + n] = acc[v];
  }
}

// ---------------- GCN aggregation pieces ----------------

__global__ void k_scale(const float* __restrict__ H, const float* __restrict__ dinv,
                        float* __restrict__ HS, int Wd, long total) {
  long idx = (long)blockIdx.x * blockDim.x + threadIdx.x;
  if (idx < total) {
    long i = idx / Wd;
    HS[idx] = H[idx] * dinv[i];
  }
}

__global__ void k_scatter(const long long* __restrict__ src, const long long* __restrict__ dst,
                          const float* __restrict__ HS, float* __restrict__ ACC,
                          int Wd, long total) {
  long idx = (long)blockIdx.x * blockDim.x + threadIdx.x;
  if (idx < total) {
    long e = idx / Wd;
    int  j = (int)(idx - e * Wd);
    long s = (long)src[e];
    long d = (long)dst[e];
    atomicAdd(&ACC[d * Wd + j], HS[s * Wd + j]);   // L2-resident f32 atomic
  }
}

__global__ void k_finalize(float* __restrict__ ACC, const float* __restrict__ HS,
                           const float* __restrict__ dinv, const float* __restrict__ bias,
                           int Wd, long total) {
  long idx = (long)blockIdx.x * blockDim.x + threadIdx.x;
  if (idx < total) {
    long i = idx / Wd;
    int  j = (int)(idx - i * Wd);
    float v = (ACC[idx] + HS[idx]) * dinv[i] + bias[j];
    ACC[idx] = v > 0.0f ? v : 0.0f;                // ReLU, in place
  }
}

// ---------------- pooling + head ----------------

__global__ void k_pool_sum(const float* __restrict__ Xf, const long long* __restrict__ batch,
                           float* __restrict__ gsum, long total) {
  long idx = (long)blockIdx.x * blockDim.x + threadIdx.x;
  if (idx < total) {
    long i = idx >> 7;           // /128
    int  j = (int)(idx & 127);
    atomicAdd(&gsum[(long)batch[i] * 128 + j], Xf[idx]);
  }
}

__global__ void k_pool_cnt(const long long* __restrict__ batch, float* __restrict__ gcnt, int n) {
  int i = blockIdx.x * blockDim.x + threadIdx.x;
  if (i < n) atomicAdd(&gcnt[(long)batch[i]], 1.0f);
}

__global__ void k_pool_div(const float* __restrict__ gsum, const float* __restrict__ gcnt,
                           float* __restrict__ g, int total) {
  int idx = blockIdx.x * blockDim.x + threadIdx.x;
  if (idx < total) {
    int gi = idx >> 7;
    g[idx] = gsum[idx] / fmaxf(gcnt[gi], 1.0f);
  }
}

// one wave per graph row; lanes 16-31 mirror 0-15 so 32-wide xor reductions are safe
__global__ void k_logsoftmax(const float* __restrict__ logits, const float* __restrict__ bl,
                             float* __restrict__ out, int G) {
  int lane = threadIdx.x & 31;
  int wave = threadIdx.x >> 5;
  int row  = blockIdx.x * (blockDim.x >> 5) + wave;
  if (row >= G) return;
  int c = lane & 15;
  float v = logits[row * NC + c] + bl[c];
  float mx = v;
#pragma unroll
  for (int m = 1; m < 16; m <<= 1) mx = fmaxf(mx, __shfl_xor(mx, m, 32));
  float s = __expf(v - mx);
#pragma unroll
  for (int m = 1; m < 16; m <<= 1) s += __shfl_xor(s, m, 32);
  float r = v - mx - __logf(s);
  if (lane < 16) out[row * NC + c] = r;
}

// ---------------- driver ----------------

extern "C" void kernel_launch(void* const* d_in, const int* in_sizes, int n_in,
                              void* d_out, int out_size, void* d_ws, size_t ws_size,
                              hipStream_t stream) {
  const float*     x     = (const float*)d_in[0];
  const long long* ei    = (const long long*)d_in[1];     // [2, NE] int64
  const long long* srcI  = ei;
  const long long* dstI  = ei + NE;
  const long long* batch = (const long long*)d_in[2];
  const float* W1 = (const float*)d_in[3];  const float* b1 = (const float*)d_in[4];
  const float* W2 = (const float*)d_in[5];  const float* b2 = (const float*)d_in[6];
  const float* W3 = (const float*)d_in[7];  const float* b3 = (const float*)d_in[8];
  const float* Wl = (const float*)d_in[9];  const float* bl = (const float*)d_in[10];
  float* out = (float*)d_out;

  float* ws    = (float*)d_ws;
  float* deg   = ws;                         // NN
  float* dinv  = deg  + NN;                  // NN
  float* bufA  = dinv + NN;                  // NN*128
  float* bufB  = bufA + (long)NN * 128;      // NN*128
  float* gsum  = bufB + (long)NN * 128;      // NG*128
  float* gcnt  = gsum + NG * 128;            // NG
  float* gmean = gcnt + NG;                  // NG*128
  float* logit = gmean + NG * 128;           // NG*NC

  const int T = 256;                         // 8 wave32 per block

  // degrees (self loop contributes 1.0 baseline)
  k_fill    <<<cdiv_h(NN, T), T, 0, stream>>>(deg, 1.0f, NN);
  k_edge_deg<<<cdiv_h(NE, T), T, 0, stream>>>(dstI, deg, NE);
  k_dinv    <<<cdiv_h(NN, T), T, 0, stream>>>(deg, dinv, NN);

  // ---- layer 1: x(NNx13) @ W1(13x16) ; H->bufA, HS->bufB, out in bufA ----
  {
    const int No = 16;
    dim3 g(cdiv_h(NN / 16, 8), No / 16);
    k_gemm_wmma<13, 16><<<g, T, 0, stream>>>(x, W1, bufA, NN);
    long tot = (long)NN * No, etot = (long)NE * No;
    k_scale   <<<cdiv_h(tot,  T), T, 0, stream>>>(bufA, dinv, bufB, No, tot);
    k_fill    <<<cdiv_h(tot,  T), T, 0, stream>>>(bufA, 0.0f, tot);
    k_scatter <<<cdiv_h(etot, T), T, 0, stream>>>(srcI, dstI, bufB, bufA, No, etot);
    k_finalize<<<cdiv_h(tot,  T), T, 0, stream>>>(bufA, bufB, dinv, b1, No, tot);
  }
  // ---- layer 2: bufA(NNx16) @ W2(16x64) ; H->bufB, HS->bufA, out in bufB ----
  {
    const int No = 64;
    dim3 g(cdiv_h(NN / 16, 8), No / 16);
    k_gemm_wmma<16, 64><<<g, T, 0, stream>>>(bufA, W2, bufB, NN);
    long tot = (long)NN * No, etot = (long)NE * No;
    k_scale   <<<cdiv_h(tot,  T), T, 0, stream>>>(bufB, dinv, bufA, No, tot);
    k_fill    <<<cdiv_h(tot,  T), T, 0, stream>>>(bufB, 0.0f, tot);
    k_scatter <<<cdiv_h(etot, T), T, 0, stream>>>(srcI, dstI, bufA, bufB, No, etot);
    k_finalize<<<cdiv_h(tot,  T), T, 0, stream>>>(bufB, bufA, dinv, b2, No, tot);
  }
  // ---- layer 3: bufB(NNx64) @ W3(64x128) ; H->bufA, HS->bufB, out in bufA ----
  {
    const int No = 128;
    dim3 g(cdiv_h(NN / 16, 8), No / 16);
    k_gemm_wmma<64, 128><<<g, T, 0, stream>>>(bufB, W3, bufA, NN);
    long tot = (long)NN * No, etot = (long)NE * No;
    k_scale   <<<cdiv_h(tot,  T), T, 0, stream>>>(bufA, dinv, bufB, No, tot);
    k_fill    <<<cdiv_h(tot,  T), T, 0, stream>>>(bufA, 0.0f, tot);
    k_scatter <<<cdiv_h(etot, T), T, 0, stream>>>(srcI, dstI, bufB, bufA, No, etot);
    k_finalize<<<cdiv_h(tot,  T), T, 0, stream>>>(bufA, bufB, dinv, b3, No, tot);
  }

  // ---- global mean pool over 512 graphs (bufA is NNx128) ----
  k_fill    <<<cdiv_h(NG * 128, T), T, 0, stream>>>(gsum, 0.0f, NG * 128);
  k_fill    <<<cdiv_h(NG,       T), T, 0, stream>>>(gcnt, 0.0f, NG);
  k_pool_sum<<<cdiv_h((long)NN * 128, T), T, 0, stream>>>(bufA, batch, gsum, (long)NN * 128);
  k_pool_cnt<<<cdiv_h(NN, T), T, 0, stream>>>(batch, gcnt, NN);
  k_pool_div<<<cdiv_h(NG * 128, T), T, 0, stream>>>(gsum, gcnt, gmean, NG * 128);

  // ---- head: gmean(512x128) @ Wl(128x16), +bl, log_softmax ----
  {
    dim3 g(cdiv_h(NG / 16, 8), 1);
    k_gemm_wmma<128, 16><<<g, T, 0, stream>>>(gmean, Wl, logit, NG);
  }
  k_logsoftmax<<<cdiv_h((long)NG * 32, T), T, 0, stream>>>(logit, bl, out, NG);
}
